// PointNetSetAbstraction_60507499266140
// MI455X (gfx1250) — compile-verified, matrix operand order checked by reading, not run on hardware
//
#include <hip/hip_runtime.h>

typedef __attribute__((ext_vector_type(16))) _Float16 v16h;
typedef __attribute__((ext_vector_type(8)))  float    v8f;

#define NPOINT  1024
#define NSAMPLE 32
#define BATCH   16
#define NPTS    4096
#define DFEAT   64
#define RADIUS2 0.04f
#define MROWS   524288L   /* BATCH*NPOINT*NSAMPLE */

// ---------------------------------------------------------------- utilities
__global__ void zero_stats_kernel(float* __restrict__ stats, int n) {
    int i = blockIdx.x * blockDim.x + threadIdx.x;
    if (i < n) stats[i] = 0.f;
}

// Pre-convert f32 weights (Cout,Cin) -> zero-padded f16 (Cout,CinPad), done once.
template <int Cin, int CinPad, int Cout>
__global__ void wconv_kernel(const float* __restrict__ W, _Float16* __restrict__ out) {
    int i = blockIdx.x * blockDim.x + threadIdx.x;
    if (i < Cout * CinPad) {
        int o = i / CinPad, c = i % CinPad;
        out[i] = (c < Cin) ? (_Float16)W[o * Cin + c] : (_Float16)0.f;
    }
}

// ---------------------------------------------------------------- FPS
__global__ __launch_bounds__(256)
void fps_kernel(const float* __restrict__ xyz, float* __restrict__ new_xyz) {
    __shared__ float dist[NPTS];
    __shared__ float rv[256];
    __shared__ int   ri[256];
    const int b   = blockIdx.x;
    const int tid = threadIdx.x;
    const float* xb = xyz + (size_t)b * NPTS * 3;

    for (int n = tid; n < NPTS; n += 256) dist[n] = 1e10f;
    __syncthreads();

    int far = 0;
    for (int it = 0; it < NPOINT; ++it) {
        float cx = xb[far * 3 + 0], cy = xb[far * 3 + 1], cz = xb[far * 3 + 2];
        if (tid == 0) {
            float* o = new_xyz + ((size_t)b * NPOINT + it) * 3;
            o[0] = cx; o[1] = cy; o[2] = cz;
        }
        float bv = -1.f; int bi = 0;
        for (int n = tid; n < NPTS; n += 256) {
            float dx = xb[n * 3 + 0] - cx;
            float dy = xb[n * 3 + 1] - cy;
            float dz = xb[n * 3 + 2] - cz;
            float d  = dx * dx + dy * dy + dz * dz;
            float dm = fminf(dist[n], d);
            dist[n] = dm;
            if (dm > bv) { bv = dm; bi = n; }
        }
        rv[tid] = bv; ri[tid] = bi;
        __syncthreads();
        for (int s = 128; s > 0; s >>= 1) {
            if (tid < s) {
                float ov = rv[tid + s]; int oi = ri[tid + s];
                if (ov > rv[tid] || (ov == rv[tid] && oi < ri[tid])) { rv[tid] = ov; ri[tid] = oi; }
            }
            __syncthreads();
        }
        far = ri[0];
        __syncthreads();
    }
}

// ---------------------------------------------------------------- ball query + gather
__global__ __launch_bounds__(256)
void group_kernel(const float* __restrict__ xyz, const float* __restrict__ points,
                  const float* __restrict__ new_xyz, _Float16* __restrict__ feat) {
    __shared__ int slist[8][NSAMPLE];
    const int wv   = threadIdx.x >> 5;
    const int lane = threadIdx.x & 31;
    const int gid  = blockIdx.x * 8 + wv;   // 0..16383
    const int b    = gid >> 10;
    const float* xb = xyz + (size_t)b * NPTS * 3;
    const float qx = new_xyz[gid * 3 + 0];
    const float qy = new_xyz[gid * 3 + 1];
    const float qz = new_xyz[gid * 3 + 2];

    int cnt = 0;
    for (int base = 0; base < NPTS; base += 32) {
        int n = base + lane;
        float dx = xb[n * 3 + 0] - qx;
        float dy = xb[n * 3 + 1] - qy;
        float dz = xb[n * 3 + 2] - qz;
        float d  = dx * dx + dy * dy + dz * dz;
        bool in  = (d <= RADIUS2);
        unsigned long long m = __ballot(in);
        int prefix = (int)__popcll(m & ((1ull << lane) - 1ull));
        int pos = cnt + prefix;
        if (in && pos < NSAMPLE) slist[wv][pos] = n;
        cnt += (int)__popcll(m);
        if (cnt >= NSAMPLE) break;
    }
    __builtin_amdgcn_wave_barrier();
    __syncthreads();

    int gi;
    if (cnt == 0) {
        gi = NPTS - 1;                       // min(N, N-1) fallback
    } else {
        int first = slist[wv][0];
        int ccl = cnt < NSAMPLE ? cnt : NSAMPLE;
        gi = (lane < ccl) ? slist[wv][lane] : first;
    }

    const long row = (long)gid * NSAMPLE + lane;
    _Float16* fr = feat + row * 96;
    fr[0] = (_Float16)(xb[gi * 3 + 0] - qx);
    fr[1] = (_Float16)(xb[gi * 3 + 1] - qy);
    fr[2] = (_Float16)(xb[gi * 3 + 2] - qz);
    const float* pb = points + ((size_t)b * NPTS + gi) * DFEAT;
#pragma unroll
    for (int c = 0; c < DFEAT; ++c) fr[3 + c] = (_Float16)pb[c];
#pragma unroll
    for (int c = 3 + DFEAT; c < 96; ++c) fr[c] = (_Float16)0.f;
}

// ---------------------------------------------------------------- WMMA GEMM
// out[m,n] = sum_c bnrelu(A[m,c]) * Wh[n,c] + bias[n], f16 in / f32 accumulate.
//  * weight panel pulled into LDS with CDNA5 async-to-LDS (ASYNCcnt) DMA
//  * optional folded BN+ReLU applied on A fragments at load (coefIn in LDS)
//  * per-channel sum/sumsq accumulated for the NEXT layer's batch-norm
template <int CinPad, int Cout, bool BN>
__global__ __launch_bounds__(256)
void gemm_kernel(const _Float16* __restrict__ A, const _Float16* __restrict__ Wh,
                 const float* __restrict__ bias, const float* __restrict__ coefIn,
                 _Float16* __restrict__ X,
                 float* __restrict__ gsum, float* __restrict__ gssq) {
    constexpr int KC = CinPad / 32;
    constexpr int NT = Cout / 16;
    constexpr int WBYTES = Cout * CinPad * 2;
    constexpr int ITER = WBYTES / (256 * 16);     // exact for all three layers
    __shared__ alignas(128) _Float16 wl[Cout * CinPad];
    __shared__ float ccoef[BN ? 2 * CinPad : 1];
    __shared__ float ssum[Cout];
    __shared__ float ssq[Cout];

    const int tid = threadIdx.x;

    // -------- async DMA: weight panel global -> LDS (16B per lane per iter)
    {
        const unsigned ldsbase = (unsigned)(size_t)(&wl[0]);
        const char* g = (const char*)Wh;
#pragma unroll
        for (int it = 0; it < ITER; ++it) {
            const int off = (it * 256 + tid) * 16;
            unsigned l = ldsbase + (unsigned)off;
            unsigned long long ga = (unsigned long long)(size_t)(g + off);
            asm volatile("global_load_async_to_lds_b128 %0, %1, off"
                         :: "v"(l), "v"(ga) : "memory");
        }
    }

    if (BN) {
        for (int i = tid; i < 2 * CinPad; i += 256) ccoef[i] = coefIn[i];
    }
    for (int i = tid; i < Cout; i += 256) { ssum[i] = 0.f; ssq[i] = 0.f; }

    asm volatile("s_wait_asynccnt 0" ::: "memory");
    __syncthreads();

    const int wave = tid >> 5, lane = tid & 31;
    const int lm = lane & 15, lh = lane >> 4;
    const long rowbase = ((long)blockIdx.x * 8 + wave) * 16;

    v16h afr[KC];
    const _Float16* arow = A + (rowbase + lm) * CinPad + lh * 16;
#pragma unroll
    for (int kc = 0; kc < KC; ++kc) {
        v16h a = *(const v16h*)(arow + kc * 32);
        if (BN) {
            const int cbase = kc * 32 + lh * 16;
#pragma unroll
            for (int j = 0; j < 16; ++j) {
                float y = ccoef[cbase + j] * (float)a[j] + ccoef[CinPad + cbase + j];
                a[j] = (_Float16)fmaxf(y, 0.f);
            }
        }
        afr[kc] = a;
    }

#pragma unroll
    for (int nt = 0; nt < NT; ++nt) {
        v8f acc = {};
        const _Float16* brow = &wl[(nt * 16 + lm) * CinPad + lh * 16];
#pragma unroll
        for (int kc = 0; kc < KC; ++kc) {
            v16h bfr = *(const v16h*)(brow + kc * 32);
            acc = __builtin_amdgcn_wmma_f32_16x16x32_f16(
                false, afr[kc], false, bfr, (short)0, acc, false, false);
        }
        const int n = nt * 16 + lm;
        const float bn = bias[n];
        // single base address; per-row byte offsets are compile-time constants
        _Float16* xp = X + (rowbase + 8 * lh) * Cout + n;
        float s = 0.f, q = 0.f;
#pragma unroll
        for (int i = 0; i < 8; ++i) {
            float v = acc[i] + bn;
            xp[i * Cout] = (_Float16)v;               // global_store_b16 offset:i*Cout*2
            s += v; q += v * v;
        }
        atomicAdd(&ssum[n], s);
        atomicAdd(&ssq[n], q);
    }
    __syncthreads();
    for (int i = tid; i < Cout; i += 256) {
        atomicAdd(&gsum[i], ssum[i]);
        atomicAdd(&gssq[i], ssq[i]);
    }
}

// ---------------------------------------------------------------- BN coefficient fold
__global__ void bn_coef_kernel(const float* __restrict__ sum, const float* __restrict__ ssq,
                               const float* __restrict__ g, const float* __restrict__ bt,
                               float* __restrict__ coef, int Cout, float invM) {
    int c = threadIdx.x;
    if (c < Cout) {
        float mean = sum[c] * invM;
        float var  = ssq[c] * invM - mean * mean;
        float a = g[c] * rsqrtf(var + 1e-5f);
        coef[c]        = a;
        coef[Cout + c] = bt[c] - mean * a;
    }
}

// ---------------------------------------------------------------- layer3 BN + ReLU + max over K
__global__ __launch_bounds__(256)
void bn_relu_max_kernel(const _Float16* __restrict__ X, const float* __restrict__ coef,
                        float* __restrict__ out) {
    int idx = blockIdx.x * 256 + threadIdx.x;   // < 16384*128
    int c  = idx & 127;
    long bs = idx >> 7;
    float a = coef[c], b2 = coef[128 + c];
    float m = 0.f;                               // relu outputs are >= 0
#pragma unroll 4
    for (int k = 0; k < NSAMPLE; ++k) {
        float x = (float)X[(bs * NSAMPLE + k) * 128 + c];
        m = fmaxf(m, fmaxf(a * x + b2, 0.f));
    }
    out[idx] = m;
}

// ---------------------------------------------------------------- launch
extern "C" void kernel_launch(void* const* d_in, const int* in_sizes, int n_in,
                              void* d_out, int out_size, void* d_ws, size_t ws_size,
                              hipStream_t stream) {
    const float* xyz    = (const float*)d_in[0];
    const float* points = (const float*)d_in[1];
    const float* w1 = (const float*)d_in[2];
    const float* b1 = (const float*)d_in[3];
    const float* g1 = (const float*)d_in[4];
    const float* bt1 = (const float*)d_in[5];
    const float* w2 = (const float*)d_in[6];
    const float* b2 = (const float*)d_in[7];
    const float* g2 = (const float*)d_in[8];
    const float* bt2 = (const float*)d_in[9];
    const float* w3 = (const float*)d_in[10];
    const float* b3 = (const float*)d_in[11];
    const float* g3 = (const float*)d_in[12];
    const float* bt3 = (const float*)d_in[13];

    float* out        = (float*)d_out;
    float* new_xyz    = out;                 // 16*1024*3
    float* new_points = out + BATCH * NPOINT * 3;

    char* ws = (char*)d_ws;
    _Float16* feat0 = (_Float16*)ws;                                   // M*96 f16
    _Float16* x3    = (_Float16*)ws;                                   // M*128 (reuses dead feat0/x1)
    _Float16* x1    = (_Float16*)(ws + (size_t)MROWS * 96 * 2);        // M*64
    _Float16* x2    = (_Float16*)(ws + (size_t)MROWS * (96 + 64) * 2); // M*64
    float* stats = (float*)(ws + (size_t)MROWS * (96 + 64 + 64) * 2);
    float *sum1 = stats,        *ssq1 = stats + 128;
    float *sum2 = stats + 256,  *ssq2 = stats + 384;
    float *sum3 = stats + 512,  *ssq3 = stats + 640;
    float *coef1 = stats + 768, *coef2 = stats + 896, *coef3 = stats + 1024;
    _Float16* wf1 = (_Float16*)((char*)stats + 8192);    // 64*96
    _Float16* wf2 = wf1 + 64 * 96;                       // 64*64
    _Float16* wf3 = wf2 + 64 * 64;                       // 128*64

    const float invM = 1.0f / (float)MROWS;

    zero_stats_kernel<<<3, 256, 0, stream>>>(stats, 768);
    wconv_kernel<67, 96, 64><<<24, 256, 0, stream>>>(w1, wf1);
    wconv_kernel<64, 64, 64><<<16, 256, 0, stream>>>(w2, wf2);
    wconv_kernel<64, 64, 128><<<32, 256, 0, stream>>>(w3, wf3);

    fps_kernel<<<BATCH, 256, 0, stream>>>(xyz, new_xyz);
    group_kernel<<<2048, 256, 0, stream>>>(xyz, points, new_xyz, feat0);

    gemm_kernel<96, 64, false><<<4096, 256, 0, stream>>>(feat0, wf1, b1, nullptr, x1, sum1, ssq1);
    bn_coef_kernel<<<1, 64, 0, stream>>>(sum1, ssq1, g1, bt1, coef1, 64, invM);

    gemm_kernel<64, 64, true><<<4096, 256, 0, stream>>>(x1, wf2, b2, coef1, x2, sum2, ssq2);
    bn_coef_kernel<<<1, 64, 0, stream>>>(sum2, ssq2, g2, bt2, coef2, 64, invM);

    gemm_kernel<64, 128, true><<<4096, 256, 0, stream>>>(x2, wf3, b3, coef2, x3, sum3, ssq3);
    bn_coef_kernel<<<1, 128, 0, stream>>>(sum3, ssq3, g3, bt3, coef3, 128, invM);

    bn_relu_max_kernel<<<8192, 256, 0, stream>>>(x3, coef3, new_points);
}
